// LayoutModel_57629871178588
// MI455X (gfx1250) — matrix operand order repaired and checked by reading, not compiled, not run on hardware
//
#include <hip/hip_runtime.h>
#include <hip/hip_bf16.h>

// ---------------------------------------------------------------------------
// CDNA5 (gfx1250) implementation of the layout GNN.
// All matmuls run on v_wmma_f32_16x16x32_f16 (wave32, fp32 accum).
// Each wave register-blocks 4 row-tiles x 1 col-tile. Per K-step: all B and A
// fragments are loaded into distinct registers first (one load clause, one
// wait), then 8 WMMAs issue back-to-back with no source-overlap hazards.
// Activations live in FP16; segment-means use fp32 atomics.
// ---------------------------------------------------------------------------

typedef _Float16 f16;
typedef __attribute__((ext_vector_type(16))) _Float16 v16h;
typedef __attribute__((ext_vector_type(8)))  _Float16 v8h;
typedef __attribute__((ext_vector_type(8)))  float    v8f;

#define TPB 256
#define MT 4  // row-tiles per wave in the GEMM

// --------------------------- elementwise helpers ---------------------------

__global__ void k_zero_f32(float* __restrict__ p, long long n) {
  for (long long t = blockIdx.x * (long long)blockDim.x + threadIdx.x; t < n;
       t += (long long)gridDim.x * blockDim.x)
    p[t] = 0.f;
}

__global__ void k_deg(const int* __restrict__ dst, int nE, float* __restrict__ deg) {
  int e = blockIdx.x * blockDim.x + threadIdx.x;
  if (e < nE) atomicAdd(&deg[dst[e]], 1.0f);
}

__global__ void k_inv(const float* __restrict__ deg, float* __restrict__ inv, int n) {
  int i = blockIdx.x * blockDim.x + threadIdx.x;
  if (i < n) inv[i] = 1.0f / fmaxf(deg[i], 1.0f);
}

// x0 = l2norm(concat(node_feat, embedding[opcode])) stored fp16, K padded 172->192
__global__ void k_embed_norm(const float* __restrict__ feat, const int* __restrict__ opc,
                             const float* __restrict__ emb, f16* __restrict__ x0, int n) {
  int i = blockIdx.x * blockDim.x + threadIdx.x;
  if (i >= n) return;
  const float* f = feat + (size_t)i * 140;
  const float* e = emb + (size_t)opc[i] * 32;
  float ss = 0.f;
  for (int k = 0; k < 140; ++k) ss += f[k] * f[k];
  for (int k = 0; k < 32; ++k) ss += e[k] * e[k];
  float sc = 1.0f / fmaxf(sqrtf(ss), 1e-12f);
  f16* o = x0 + (size_t)i * 192;
  for (int k = 0; k < 140; ++k) o[k] = (f16)(f[k] * sc);
  for (int k = 0; k < 32; ++k) o[140 + k] = (f16)(e[k] * sc);
  for (int k = 172; k < 192; ++k) o[k] = (f16)0.f;
}

// Pack fp32 weight W[Korig, Ncols] into fp16 WMMA B-fragment order:
// P[((kb*NB+nb)*32 + lane)*16 + j] = W[kb*32 + (lane/16)*16 + j, nb*16 + lane%16]
__global__ void k_pack_w(const float* __restrict__ W, f16* __restrict__ P,
                         int Korig, int Kpad, int Ncols, int NB) {
  long long total = (long long)(Kpad >> 5) * NB * 512;
  for (long long t = blockIdx.x * (long long)blockDim.x + threadIdx.x; t < total;
       t += (long long)gridDim.x * blockDim.x) {
    int j = (int)(t & 15);
    int lane = (int)((t >> 4) & 31);
    long long rest = t >> 9;
    int nb = (int)(rest % NB);
    int kb = (int)(rest / NB);
    int kk = kb * 32 + (lane >> 4) * 16 + j;
    int col = nb * 16 + (lane & 15);
    float v = (kk < Korig && col < Ncols) ? W[(size_t)kk * Ncols + col] : 0.f;
    P[t] = (f16)v;
  }
}

// msg[dst] += x[src] per edge per feature (fp32 atomics); batched for cfg graphs
__global__ void k_scatter(const f16* __restrict__ x, int stride,
                          const int* __restrict__ src, const int* __restrict__ dst,
                          int nE, int nodesPerBatch, int batches,
                          float* __restrict__ msg) {
  long long total = (long long)batches * nE * stride;
  for (long long t = blockIdx.x * (long long)blockDim.x + threadIdx.x; t < total;
       t += (long long)gridDim.x * blockDim.x) {
    int k = (int)(t % stride);
    long long e2 = t / stride;
    int e = (int)(e2 % nE);
    int b = (int)(e2 / nE);
    long long s = (long long)b * nodesPerBatch + src[e];
    long long d = (long long)b * nodesPerBatch + dst[e];
    float v = (float)x[s * stride + k];
    atomicAdd(&msg[d * stride + k], v);
  }
}

// msg_h = (f16)(msg * inv[node % invmod])
__global__ void k_scale_h(const float* __restrict__ msg, const float* __restrict__ inv,
                          int invmod, f16* __restrict__ out, long long total, int stride) {
  for (long long t = blockIdx.x * (long long)blockDim.x + threadIdx.x; t < total;
       t += (long long)gridDim.x * blockDim.x) {
    long long node = t / stride;
    out[t] = (f16)(msg[t] * inv[(int)(node % invmod)]);
  }
}

// ------------------------------ WMMA GEMM ----------------------------------
// Out[M,64] = act( A1@W1 (+ A2@W2) (+ bias) ), fp16 in/out, fp32 accum.
// Wave handles MT row-tiles x one 16-col tile. Per K-step: load everything
// into distinct regs, then issue all WMMAs as one dense cluster.
__global__ __launch_bounds__(TPB) void k_gemm_wmma(
    const f16* __restrict__ A1, const f16* __restrict__ W1,
    const f16* __restrict__ A2, const f16* __restrict__ W2,
    const float* __restrict__ bias, f16* __restrict__ Out,
    int M, int Mreal, int Kblocks, int Astride, int NB, int outStride, int doLrelu) {
  int wave = blockIdx.x * (blockDim.x >> 5) + (threadIdx.x >> 5);
  int lane = threadIdx.x & 31;
  int rowTiles = M >> 4;
  int rowGroups = (rowTiles + MT - 1) / MT;
  int tiles = rowGroups * NB;
  if (wave >= tiles) return;  // wave-uniform: EXEC stays all-ones for WMMA
  int rg = wave / NB, nb = wave % NB;
  int rt0 = rg * MT;

  // Per-row-tile clamped A row index (loads clamped, stores guarded)
  int rcl[MT];
#pragma unroll
  for (int mt = 0; mt < MT; ++mt) {
    int rowA = (rt0 + mt) * 16 + (lane & 15);
    rcl[mt] = rowA < Mreal ? rowA : 0;
  }
  int kofs = (lane >> 4) * 16;  // A layout: lanes 0-15 K 0-15, 16-31 K 16-31

  v8f acc[MT];
#pragma unroll
  for (int mt = 0; mt < MT; ++mt) acc[mt] = (v8f){};

  union AV { v16h v; v8h h[2]; };
  if (A2) {
    // dual-stream SAGE form: msg@Wl + x@Wr
    AV a1[MT], a2[MT];
    for (int kb = 0; kb < Kblocks; ++kb) {
      v16h b1 = *(const v16h*)(W1 + (((size_t)kb * NB + nb) * 32 + lane) * 16);
      v16h b2 = *(const v16h*)(W2 + (((size_t)kb * NB + nb) * 32 + lane) * 16);
#pragma unroll
      for (int mt = 0; mt < MT; ++mt) {
        const f16* p = A1 + (size_t)rcl[mt] * Astride + kb * 32;
        a1[mt].h[0] = *(const v8h*)(p + kofs);
        a1[mt].h[1] = *(const v8h*)(p + kofs + 8);
      }
#pragma unroll
      for (int mt = 0; mt < MT; ++mt) {
        const f16* p = A2 + (size_t)rcl[mt] * Astride + kb * 32;
        a2[mt].h[0] = *(const v8h*)(p + kofs);
        a2[mt].h[1] = *(const v8h*)(p + kofs + 8);
      }
#pragma unroll
      for (int mt = 0; mt < MT; ++mt)
        acc[mt] = __builtin_amdgcn_wmma_f32_16x16x32_f16(false, a1[mt].v, false, b1,
                                                         (short)0, acc[mt], false, false);
#pragma unroll
      for (int mt = 0; mt < MT; ++mt)
        acc[mt] = __builtin_amdgcn_wmma_f32_16x16x32_f16(false, a2[mt].v, false, b2,
                                                         (short)0, acc[mt], false, false);
    }
  } else {
    // single-stream: plain GEMM (+bias)
    AV a1[MT];
    for (int kb = 0; kb < Kblocks; ++kb) {
      v16h b1 = *(const v16h*)(W1 + (((size_t)kb * NB + nb) * 32 + lane) * 16);
#pragma unroll
      for (int mt = 0; mt < MT; ++mt) {
        const f16* p = A1 + (size_t)rcl[mt] * Astride + kb * 32;
        a1[mt].h[0] = *(const v8h*)(p + kofs);
        a1[mt].h[1] = *(const v8h*)(p + kofs + 8);
      }
#pragma unroll
      for (int mt = 0; mt < MT; ++mt)
        acc[mt] = __builtin_amdgcn_wmma_f32_16x16x32_f16(false, a1[mt].v, false, b1,
                                                         (short)0, acc[mt], false, false);
    }
  }

  int col = nb * 16 + (lane & 15);
  float bv = bias ? bias[col] : 0.f;
#pragma unroll
  for (int mt = 0; mt < MT; ++mt) {
    int rbase = (rt0 + mt) * 16 + (lane >> 4) * 8;  // D layout: vgpr r -> row 8*(lane/16)+r
    for (int r = 0; r < 8; ++r) {
      float v = acc[mt][r] + bv;
      if (doLrelu) v = v > 0.f ? v : 0.01f * v;
      int row = rbase + r;
      if (row < Mreal) Out[(size_t)row * outStride + col] = (f16)v;
    }
  }
}

// out[i,:] = src[ids[i],:] for i<nReal, else 0 (stride-wide rows)
__global__ void k_gather_rows(const f16* __restrict__ src, int stride,
                              const int* __restrict__ ids, f16* __restrict__ out,
                              int nOut, int nReal) {
  long long total = (long long)nOut * stride;
  for (long long t = blockIdx.x * (long long)blockDim.x + threadIdx.x; t < total;
       t += (long long)gridDim.x * blockDim.x) {
    int row = (int)(t / stride);
    int k = (int)(t % stride);
    out[t] = (row < nReal) ? src[(size_t)ids[row] * stride + k] : (f16)0.f;
  }
}

// pad node_config_feat [C,NC,18] -> fp16 A [C*NCp, 32]
__global__ void k_pad_cfg(const float* __restrict__ ncf, f16* __restrict__ A,
                          int NCr, int NCp, int C) {
  long long total = (long long)C * NCp * 32;
  for (long long t = blockIdx.x * (long long)blockDim.x + threadIdx.x; t < total;
       t += (long long)gridDim.x * blockDim.x) {
    int k = (int)(t & 31);
    long long row = t >> 5;
    int c = (int)(row / NCp);
    int il = (int)(row % NCp);
    float v = (il < NCr && k < 18) ? ncf[((size_t)c * NCr + il) * 18 + k] : 0.f;
    A[t] = (f16)v;
  }
}

// h[c*NCp+i, 0:192] = l2norm(concat(nbrs[i], xfin[ids[i]], cf[c*NCp+i]))
__global__ void k_build_h(const f16* __restrict__ nbrs, const f16* __restrict__ xf,
                          const int* __restrict__ ids, const f16* __restrict__ cf,
                          f16* __restrict__ h, int NCr, int NCp, int C) {
  long long CN = (long long)C * NCp;
  long long i = blockIdx.x * (long long)blockDim.x + threadIdx.x;
  if (i >= CN) return;
  int il = (int)(i % NCp);
  f16* o = h + i * 192;
  if (il >= NCr) {
    for (int k = 0; k < 192; ++k) o[k] = (f16)0.f;
    return;
  }
  const f16* a = nbrs + (size_t)il * 64;
  const f16* b = xf + (size_t)ids[il] * 64;
  const f16* d = cf + i * 64;
  float ss = 0.f;
  for (int k = 0; k < 64; ++k) { float v = (float)a[k]; ss += v * v; }
  for (int k = 0; k < 64; ++k) { float v = (float)b[k]; ss += v * v; }
  for (int k = 0; k < 64; ++k) { float v = (float)d[k]; ss += v * v; }
  float sc = 1.0f / fmaxf(sqrtf(ss), 1e-12f);
  for (int k = 0; k < 64; ++k) o[k] = (f16)((float)a[k] * sc);
  for (int k = 0; k < 64; ++k) o[64 + k] = (f16)((float)b[k] * sc);
  for (int k = 0; k < 64; ++k) o[128 + k] = (f16)((float)d[k] * sc);
}

// pooled[c,d] = mean over i<NCr of h[c*NCp+i, d]
__global__ void k_pool(const f16* __restrict__ h, f16* __restrict__ pooled,
                       int NCr, int NCp) {
  int c = blockIdx.x;
  int d = threadIdx.x;  // 64
  const f16* base = h + ((size_t)c * NCp) * 64 + d;
  float s = 0.f;
  for (int i = 0; i < NCr; ++i) s += (float)base[(size_t)i * 64];
  pooled[c * 64 + d] = (f16)(s / (float)NCr);
}

// out[c] = t2[c,:] . W3[:,0]  (no activation on final layer)
__global__ void k_head(const f16* __restrict__ t2, const float* __restrict__ W3,
                       float* __restrict__ out) {
  int c = threadIdx.x;  // 64
  float s = 0.f;
  for (int k = 0; k < 64; ++k) s += (float)t2[c * 64 + k] * W3[k];
  out[c] = s;
}

// ------------------------------- host side ---------------------------------

static inline int blks(long long n) { return (int)((n + TPB - 1) / TPB); }

extern "C" void kernel_launch(void* const* d_in, const int* in_sizes, int n_in,
                              void* d_out, int out_size, void* d_ws, size_t ws_size,
                              hipStream_t stream) {
  (void)in_sizes; (void)n_in; (void)out_size; (void)ws_size;
  const int N = 100000, E = 400000, NC = 5000, EC = 20000, C = 64;
  const int NCp = 5008;                 // 5000 padded to multiple of 16
  const long long CN = (long long)C * NCp;

  // ---- inputs (JAX pytree / alphabetical flatten order) ----
  const float* node_feat = (const float*)d_in[0];
  const int*   node_opc  = (const int*)d_in[1];
  const int*   e_src     = (const int*)d_in[2];
  const int*   e_dst     = (const int*)d_in[2] + E;
  const float* ncfgfeat  = (const float*)d_in[3];
  const int*   cfg_ids   = (const int*)d_in[4];
  const int*   c_src     = (const int*)d_in[5];
  const int*   c_dst     = (const int*)d_in[5] + EC;
  const float* cfgWl[4]  = {(const float*)d_in[6], (const float*)d_in[9],
                            (const float*)d_in[12], (const float*)d_in[15]};
  const float* cfgWr[4]  = {(const float*)d_in[7], (const float*)d_in[10],
                            (const float*)d_in[13], (const float*)d_in[16]};
  const float* cfgBl[4]  = {(const float*)d_in[8], (const float*)d_in[11],
                            (const float*)d_in[14], (const float*)d_in[17]};
  const float* dW1 = (const float*)d_in[18];
  const float* dW2 = (const float*)d_in[19];
  const float* dW3 = (const float*)d_in[20];
  const float* embp = (const float*)d_in[21];
  const float* nbrWl[2] = {(const float*)d_in[22], (const float*)d_in[25]};
  const float* nbrWr[2] = {(const float*)d_in[23], (const float*)d_in[26]};
  const float* nbrBl[2] = {(const float*)d_in[24], (const float*)d_in[27]};
  const float* nodWl[4] = {(const float*)d_in[28], (const float*)d_in[31],
                           (const float*)d_in[34], (const float*)d_in[37]};
  const float* nodWr[4] = {(const float*)d_in[29], (const float*)d_in[32],
                           (const float*)d_in[35], (const float*)d_in[38]};
  const float* nodBl[4] = {(const float*)d_in[30], (const float*)d_in[33],
                           (const float*)d_in[36], (const float*)d_in[39]};
  const float* prjW = (const float*)d_in[40];
  const float* prjB = (const float*)d_in[41];

  // ---- workspace bump allocator ----
  size_t off = 0;
  auto alloc = [&](size_t bytes) -> char* {
    char* p = (char*)d_ws + off;
    off += bytes;
    off = (off + 255) & ~(size_t)255;
    return p;
  };
  float* deg   = (float*)alloc((size_t)N * 4);
  float* inv   = (float*)alloc((size_t)N * 4);
  float* cdeg  = (float*)alloc((size_t)NCp * 4);
  float* cinv  = (float*)alloc((size_t)NCp * 4);
  f16*   x0    = (f16*)alloc((size_t)N * 192 * 2);
  f16*   xA    = (f16*)alloc((size_t)N * 64 * 2);
  f16*   xB    = (f16*)alloc((size_t)N * 64 * 2);
  float* msgN  = (float*)alloc((size_t)N * 192 * 4);
  f16*   msgNh = (f16*)alloc((size_t)N * 192 * 2);
  f16*   nbrA  = (f16*)alloc((size_t)NCp * 64 * 2);
  f16*   nbrB  = (f16*)alloc((size_t)NCp * 64 * 2);
  float* msgC  = (float*)alloc((size_t)NCp * 64 * 4);
  f16*   msgCh = (f16*)alloc((size_t)NCp * 64 * 2);
  f16*   cfA   = (f16*)alloc((size_t)CN * 32 * 2);
  f16*   cfH   = (f16*)alloc((size_t)CN * 64 * 2);
  f16*   hA    = (f16*)alloc((size_t)CN * 192 * 2);
  f16*   hB    = (f16*)alloc((size_t)CN * 192 * 2);
  float* msgH  = (float*)alloc((size_t)CN * 192 * 4);
  f16*   msgHh = (f16*)alloc((size_t)CN * 192 * 2);
  f16*   pooled= (f16*)alloc(64 * 64 * 2);
  f16*   t1    = (f16*)alloc(64 * 64 * 2);
  f16*   t2    = (f16*)alloc(64 * 64 * 2);
  // packed weights (fp16 B-fragments)
  auto packsz = [](int Kpad, int NB) { return (size_t)(Kpad / 32) * NB * 512; };
  f16* nodWlP[4]; f16* nodWrP[4]; f16* nbrWlP[2]; f16* nbrWrP[2];
  f16* cfgWlP[4]; f16* cfgWrP[4];
  for (int l = 0; l < 4; ++l) {
    int Kp = l == 0 ? 192 : 64;
    nodWlP[l] = (f16*)alloc(packsz(Kp, 4) * 2);
    nodWrP[l] = (f16*)alloc(packsz(Kp, 4) * 2);
  }
  for (int l = 0; l < 2; ++l) {
    nbrWlP[l] = (f16*)alloc(packsz(64, 4) * 2);
    nbrWrP[l] = (f16*)alloc(packsz(64, 4) * 2);
  }
  for (int l = 0; l < 4; ++l) {
    int Kp = l == 0 ? 192 : 64;
    cfgWlP[l] = (f16*)alloc(packsz(Kp, 4) * 2);
    cfgWrP[l] = (f16*)alloc(packsz(Kp, 4) * 2);
  }
  f16* prjP = (f16*)alloc(packsz(32, 4) * 2);
  f16* dW1P = (f16*)alloc(packsz(64, 4) * 2);
  f16* dW2P = (f16*)alloc(packsz(64, 4) * 2);

  // ---- launch helpers ----
  auto zero = [&](float* p, long long n) { k_zero_f32<<<blks(n), TPB, 0, stream>>>(p, n); };
  auto pack = [&](const float* W, f16* P, int Ko, int Kp, int Nc, int NB) {
    long long t = (long long)(Kp / 32) * NB * 512;
    k_pack_w<<<blks(t), TPB, 0, stream>>>(W, P, Ko, Kp, Nc, NB);
  };
  auto gemm = [&](const f16* A1, const f16* W1, const f16* A2, const f16* W2,
                  const float* bias, f16* Out, int M, int Mreal, int Kb, int Ast,
                  int lrelu) {
    int rowGroups = (M / 16 + MT - 1) / MT;
    int tiles = rowGroups * 4;  // NB = 4
    k_gemm_wmma<<<(tiles + 7) / 8, TPB, 0, stream>>>(A1, W1, A2, W2, bias, Out, M, Mreal,
                                                     Kb, Ast, 4, 64, lrelu);
  };

  // ---- degrees ----
  zero(deg, N); zero(cdeg, NCp);
  k_deg<<<blks(E), TPB, 0, stream>>>(e_dst, E, deg);
  k_deg<<<blks(EC), TPB, 0, stream>>>(c_dst, EC, cdeg);
  k_inv<<<blks(N), TPB, 0, stream>>>(deg, inv, N);
  k_inv<<<blks(NCp), TPB, 0, stream>>>(cdeg, cinv, NCp);

  // ---- weight packing ----
  for (int l = 0; l < 4; ++l) {
    int Ko = l == 0 ? 172 : 64, Kp = l == 0 ? 192 : 64;
    pack(nodWl[l], nodWlP[l], Ko, Kp, 64, 4);
    pack(nodWr[l], nodWrP[l], Ko, Kp, 64, 4);
  }
  for (int l = 0; l < 2; ++l) {
    pack(nbrWl[l], nbrWlP[l], 64, 64, 64, 4);
    pack(nbrWr[l], nbrWrP[l], 64, 64, 64, 4);
  }
  for (int l = 0; l < 4; ++l) {
    int Ko = l == 0 ? 192 : 64, Kp = Ko;
    pack(cfgWl[l], cfgWlP[l], Ko, Kp, 64, 4);
    pack(cfgWr[l], cfgWrP[l], Ko, Kp, 64, 4);
  }
  pack(prjW, prjP, 18, 32, 64, 4);
  pack(dW1, dW1P, 64, 64, 64, 4);
  pack(dW2, dW2P, 64, 64, 64, 4);

  // ---- x0 = l2norm(concat(feat, emb)) ----
  k_embed_norm<<<blks(N), TPB, 0, stream>>>(node_feat, node_opc, embp, x0, N);

  // ---- node_gnn: 4 SAGE layers ----
  f16* xcur = x0; f16* xout[4] = {xA, xB, xA, xB};
  for (int l = 0; l < 4; ++l) {
    int K = l == 0 ? 192 : 64;
    long long msgsz = (long long)N * K;
    zero(msgN, msgsz);
    k_scatter<<<blks((long long)E * K), TPB, 0, stream>>>(xcur, K, e_src, e_dst, E, N, 1, msgN);
    k_scale_h<<<blks(msgsz), TPB, 0, stream>>>(msgN, inv, N, msgNh, msgsz, K);
    gemm(msgNh, nodWlP[l], xcur, nodWrP[l], nodBl[l], xout[l], N, N, K / 32, K, 1);
    xcur = xout[l];
  }
  f16* x4 = xcur;  // [N,64]

  // ---- nbrs0 = seg_mean(x4)[cfg_ids] ----
  {
    long long msgsz = (long long)N * 64;
    zero(msgN, msgsz);
    k_scatter<<<blks((long long)E * 64), TPB, 0, stream>>>(x4, 64, e_src, e_dst, E, N, 1, msgN);
    k_scale_h<<<blks(msgsz), TPB, 0, stream>>>(msgN, inv, N, msgNh, msgsz, 64);
    k_gather_rows<<<blks((long long)NCp * 64), TPB, 0, stream>>>(msgNh, 64, cfg_ids, nbrA,
                                                                 NCp, NC);
  }

  // ---- nbr_gnn: 2 SAGE layers on config graph ----
  f16* ncur = nbrA; f16* nout[2] = {nbrB, nbrA};
  for (int l = 0; l < 2; ++l) {
    long long msgsz = (long long)NCp * 64;
    zero(msgC, msgsz);
    k_scatter<<<blks((long long)EC * 64), TPB, 0, stream>>>(ncur, 64, c_src, c_dst, EC, NCp,
                                                            1, msgC);
    k_scale_h<<<blks(msgsz), TPB, 0, stream>>>(msgC, cinv, NCp, msgCh, msgsz, 64);
    gemm(msgCh, nbrWlP[l], ncur, nbrWrP[l], nbrBl[l], nout[l], NCp, NC, 2, 64, 1);
    ncur = nout[l];
  }

  // ---- cf = lrelu(node_config_feat @ prj_W + prj_b) ----
  k_pad_cfg<<<blks(CN * 32), TPB, 0, stream>>>(ncfgfeat, cfA, NC, NCp, C);
  gemm(cfA, prjP, nullptr, nullptr, prjB, cfH, (int)CN, (int)CN, 1, 32, 1);

  // ---- h = l2norm(concat(nbrs, xc, cf)) ----
  k_build_h<<<blks(CN), TPB, 0, stream>>>(ncur, x4, cfg_ids, cfH, hA, NC, NCp, C);

  // ---- cfg_gnn: 4 SAGE layers batched over C configs ----
  f16* hcur = hA; f16* hout[4] = {hB, hA, hB, hA};
  for (int l = 0; l < 4; ++l) {
    int K = l == 0 ? 192 : 64;
    long long msgsz = CN * K;
    zero(msgH, msgsz);
    k_scatter<<<blks((long long)C * EC * K), TPB, 0, stream>>>(hcur, K, c_src, c_dst, EC,
                                                               NCp, C, msgH);
    k_scale_h<<<blks(msgsz), TPB, 0, stream>>>(msgH, cinv, NCp, msgHh, msgsz, K);
    gemm(msgHh, cfgWlP[l], hcur, cfgWrP[l], cfgBl[l], hout[l], (int)CN, (int)CN, K / 32, K, 1);
    hcur = hout[l];
  }

  // ---- pool + dense head ----
  k_pool<<<C, 64, 0, stream>>>(hcur, pooled, NC, NCp);
  gemm(pooled, dW1P, nullptr, nullptr, nullptr, t1, 64, 64, 2, 64, 1);
  gemm(t1, dW2P, nullptr, nullptr, nullptr, t2, 64, 64, 2, 64, 1);
  k_head<<<1, 64, 0, stream>>>(t2, dW3, (float*)d_out);
}
